// MultiHeadAttention_18803366822057
// MI455X (gfx1250) — matrix operand order repaired and checked by reading, not compiled, not run on hardware
//
#include <hip/hip_runtime.h>
#include <hip/hip_bf16.h>

typedef __attribute__((ext_vector_type(16))) _Float16 v16h;
typedef __attribute__((ext_vector_type(8)))  _Float16 h8;
typedef __attribute__((ext_vector_type(8)))  float    v8f;

static constexpr int Nn = 2, Ll = 2048, Ee = 1024, Hh = 16, Dd = 64;
typedef unsigned int u32;

#define WMMA16(A, B, C) \
  __builtin_amdgcn_wmma_f32_16x16x32_f16(false, (A), false, (B), (short)0, (C), false, false)

// ---- operand loaders (per CDNA5 ISA 16-bit WMMA VGPR layouts) --------------
// All indices are 32-bit so loads/stores use the SADDR (SGPR base + 32-bit
// VGPR offset) addressing form instead of 64-bit per-lane address math.
//
// A 16x32 f16: lane l holds row m=l&15; halves 0..7 = K 8g..8g+7, halves 8..15
// = K 16+8g..16+8g+7 (g = l>>4). Both chunks are contiguous -> two b128 loads.
__device__ __forceinline__ v16h load_a_tile(const _Float16* __restrict__ base,
                                            u32 ldk, int lane) {
  const u32 m = lane & 15, g = lane >> 4;
  const u32 off = m * ldk + 8u * g;
  h8 c0 = *(const h8*)(base + off);
  h8 c1 = *(const h8*)(base + off + 16u);
  v16h a;
#pragma unroll
  for (int i = 0; i < 8; ++i) { a[i] = c0[i]; a[i + 8] = c1[i]; }
  return a;
}

// B 32x16 f16: lane l holds column n=l&15 (a source row in memory); halves
// 0..7 = K 8g..8g+7, halves 8..15 = K 16+8g..16+8g+7 -> two b128 loads.
__device__ __forceinline__ v16h load_b_tile(const _Float16* __restrict__ colbase,
                                            u32 ldk, int lane) {
  const u32 n = lane & 15, g = lane >> 4;
  const u32 off = n * ldk + 8u * g;
  h8 c0 = *(const h8*)(colbase + off);
  h8 c1 = *(const h8*)(colbase + off + 16u);
  v16h b;
#pragma unroll
  for (int i = 0; i < 8; ++i) { b[i] = c0[i]; b[i + 8] = c1[i]; }
  return b;
}

__device__ __forceinline__ float red16_max(float v) {
  v = fmaxf(v, __shfl_xor(v, 1, 32));
  v = fmaxf(v, __shfl_xor(v, 2, 32));
  v = fmaxf(v, __shfl_xor(v, 4, 32));
  v = fmaxf(v, __shfl_xor(v, 8, 32));
  return v;
}
__device__ __forceinline__ float red16_sum(float v) {
  v += __shfl_xor(v, 1, 32);
  v += __shfl_xor(v, 2, 32);
  v += __shfl_xor(v, 4, 32);
  v += __shfl_xor(v, 8, 32);
  return v;
}

// ---- kernel 1: Wo fp32 -> fp16 ---------------------------------------------
__global__ void cvt_f16_kernel(const float* __restrict__ src,
                               _Float16* __restrict__ dst, int n) {
  int i = blockIdx.x * blockDim.x + threadIdx.x;
  if (i < n) dst[i] = (_Float16)src[i];
}

// ---- kernel 2: per-head QKV projections (M=65536, K=64, N=64 GEMM) ---------
// P selects projection at compile time: 0=Q(+1/32 scale -> Qh [n,h,l,d]),
// 1=K -> Kh [n,h,l,d], 2=V -> Vt [n,h,d,l].  Compile-time strides let the
// scatter use immediate-offset SADDR stores (no 64-bit VALU address chains).
template <int P>
__global__ void __launch_bounds__(128) proj_kernel(
    const float* __restrict__ X, const float* __restrict__ W,
    _Float16* __restrict__ dstp) {
  const int lane = threadIdx.x & 31, wave = threadIdx.x >> 5;
  // softmax 1/sqrt(E) scale folded into the Q weights (B is loop-invariant)
  constexpr float scale = (P == 0) ? (1.0f / 32.0f) : 1.0f;
  // branch-free output addressing: dst[hh*(L*D) + e*se + l*sl]
  constexpr u32 se = (P == 2) ? (u32)Ll : 1u;
  constexpr u32 sl = (P == 2) ? 1u : (u32)Dd;
  const u32 n16 = lane & 15, g = lane >> 4;

  // B = W^T, resident in registers: col e = row e of W, K = d contiguous.
  v16h Breg[2][4];
#pragma unroll
  for (int c = 0; c < 4; ++c) {
    const float* wr = W + (16u * c + n16) * (u32)Dd;
#pragma unroll
    for (int s = 0; s < 2; ++s) {
      v16h b;
#pragma unroll
      for (int i = 0; i < 16; ++i)
        b[i] = (_Float16)(wr[32 * s + 8 * g + i + (i >= 8 ? 8 : 0)] * scale);
      Breg[s][c] = b;
    }
  }

  for (int it = 0; it < 8; ++it) {
    const u32 mt = (blockIdx.x * 4u + wave) * 8u + it;  // row tile 0..4095
    const u32 m0 = mt * 16u;
    // 16 consecutive rows m = (n*L + l)*H + h share one (n, l) and span all
    // 16 heads: hh = r + 8g, l/nb constant per tile.
    const u32 l  = (m0 >> 4) & (Ll - 1);
    const u32 nb = m0 >> 15;
    // fold wave-uniform parts into the destination base (stays in SGPRs);
    // lane-varying base offset handled once, r/c steps become immediates.
    _Float16* dwave = dstp + nb * (u32)(Hh * Ll * Dd) + l * sl +
                      (8u * g) * (u32)(Ll * Dd) + n16 * se;
    // A tiles (two 32-wide K slabs), fp32 -> fp16 on the fly
    v16h A[2];
    {
      const float* row = X + (m0 + n16) * (u32)Dd;
#pragma unroll
      for (int s = 0; s < 2; ++s) {
        v16h a;
#pragma unroll
        for (int i = 0; i < 8; ++i) {
          a[i]     = (_Float16)row[32 * s + 8 * g + i];
          a[i + 8] = (_Float16)row[32 * s + 16 + 8 * g + i];
        }
        A[s] = a;
      }
    }
    v8f C[4] = {};
#pragma unroll
    for (int c = 0; c < 4; ++c) {
      C[c] = WMMA16(A[0], Breg[0][c], C[c]);
      C[c] = WMMA16(A[1], Breg[1][c], C[c]);
    }
    // branch-free scatter; r/c offsets are compile-time immediates
#pragma unroll
    for (int c = 0; c < 4; ++c) {
#pragma unroll
      for (int r = 0; r < 8; ++r) {
        dwave[(u32)r * (u32)(Ll * Dd) + (u32)(16 * c) * se] =
            (_Float16)C[c][r];
      }
    }
  }
}

// ---- kernel 3: flash attention, one wave per (n, h, 16-query tile) ---------
__global__ void __launch_bounds__(128) attn_kernel(
    const _Float16* __restrict__ Qh, const _Float16* __restrict__ Kh,
    const _Float16* __restrict__ Vt, const int* __restrict__ mask,
    _Float16* __restrict__ Oc) {
  __shared__ alignas(16) _Float16 Pl[4][16][32];  // per-wave P staging (C->A)
  const int lane = threadIdx.x & 31, wave = threadIdx.x >> 5;
  const u32 tile = blockIdx.x * 4u + wave;         // 0..4095
  const u32 qt = tile & 127;
  const u32 h  = (tile >> 7) & 15;
  const u32 nb = tile >> 11;
  const u32 q0 = qt * 16u;
  const u32 g = lane >> 4, ln = lane & 15;

  const _Float16* Qb = Qh + ((nb * Hh + h) * (u32)Ll + q0) * (u32)Dd;
  const _Float16* Kb = Kh + (nb * Hh + h) * (u32)(Ll * Dd);
  const _Float16* Vb = Vt + (nb * Hh + h) * (u32)(Dd * Ll);
  const int*      Mb = mask + nb * (u32)(Ll * Ll) + q0 * (u32)Ll;

  // Q tiles resident (already scaled by 1/sqrt(E) in projection)
  const v16h qa0 = load_a_tile(Qb, Dd, lane);
  const v16h qa1 = load_a_tile(Qb + 32, Dd, lane);

  v8f O[4] = {};
  float mrun[8], lrun[8];
#pragma unroll
  for (int r = 0; r < 8; ++r) { mrun[r] = -3.0e38f; lrun[r] = 0.0f; }

  for (u32 k0 = 0; k0 < (u32)Ll; k0 += 32) {
    // S = Q K^T (two 16-key C tiles, chained over the 64-wide d dimension)
    v8f S[2];
#pragma unroll
    for (int t = 0; t < 2; ++t) {
      v8f s = {};
      s = WMMA16(qa0, load_b_tile(Kb + (k0 + 16u * t) * (u32)Dd, Dd, lane), s);
      s = WMMA16(qa1, load_b_tile(Kb + (k0 + 16u * t) * (u32)Dd + 32u, Dd, lane), s);
      const u32 kcol = k0 + 16u * t + ln;
#pragma unroll
      for (int r = 0; r < 8; ++r) {
        const u32 qrow = r + 8u * g;
        if (Mb[qrow * (u32)Ll + kcol] == 0) s[r] = -3.0e18f;
      }
      S[t] = s;
    }
    // online softmax
    float mnew[8], alpha[8];
#pragma unroll
    for (int r = 0; r < 8; ++r) {
      float tm = red16_max(fmaxf(S[0][r], S[1][r]));
      mnew[r] = fmaxf(mrun[r], tm);
      alpha[r] = __expf(mrun[r] - mnew[r]);
      mrun[r] = mnew[r];
    }
#pragma unroll
    for (int r = 0; r < 8; ++r) {
      const float p0 = __expf(S[0][r] - mnew[r]);
      const float p1 = __expf(S[1][r] - mnew[r]);
      S[0][r] = p0; S[1][r] = p1;
      lrun[r] = lrun[r] * alpha[r] + red16_sum(p0 + p1);
    }
#pragma unroll
    for (int c = 0; c < 4; ++c)
#pragma unroll
      for (int r = 0; r < 8; ++r) O[c][r] *= alpha[r];

    // re-layout P (C layout) -> A layout via per-wave LDS tile.
    // DS ops are in-order within a wave, so no barrier is required.
#pragma unroll
    for (int t = 0; t < 2; ++t)
#pragma unroll
      for (int r = 0; r < 8; ++r)
        Pl[wave][r + 8 * g][16 * t + ln] = (_Float16)S[t][r];

    v16h Pa;
    {
      const _Float16* pr = &Pl[wave][ln][0];
      h8 c0 = *(const h8*)(pr + 8 * g);
      h8 c1 = *(const h8*)(pr + 16 + 8 * g);
#pragma unroll
      for (int i = 0; i < 8; ++i) { Pa[i] = c0[i]; Pa[i + 8] = c1[i]; }
    }

    // O += P V  (B from transposed V: two 16B runs per lane)
#pragma unroll
    for (int c = 0; c < 4; ++c) {
      v16h vb = load_b_tile(Vb + (16u * c) * (u32)Ll + k0, Ll, lane);
      O[c] = WMMA16(Pa, vb, O[c]);
    }
  }

  // normalize and write concat-head output [n][q][E] as f16
  float inv[8];
#pragma unroll
  for (int r = 0; r < 8; ++r) inv[r] = 1.0f / lrun[r];
  _Float16* Ob = Oc + (nb * (u32)Ll + q0 + 8u * g) * (u32)Ee + h * (u32)Dd + ln;
#pragma unroll
  for (int c = 0; c < 4; ++c) {
#pragma unroll
    for (int r = 0; r < 8; ++r) {
      Ob[(u32)r * (u32)Ee + (u32)(16 * c)] = (_Float16)(O[c][r] * inv[r]);
    }
  }
}

// ---- kernel 4: output projection out = O @ Wo^T + bo (4096x1024x1024) ------
__global__ void __launch_bounds__(128) outproj_kernel(
    const _Float16* __restrict__ Oc, const _Float16* __restrict__ WoH,
    const float* __restrict__ bo, float* __restrict__ out) {
  const int lane = threadIdx.x & 31, wave = threadIdx.x >> 5;
  const u32 task = blockIdx.x * 4u + wave;  // 0..4095
  const u32 mt = task >> 4;                 // 256 row tiles
  const u32 ns = task & 15;                 // 16 column strips of 64
  const u32 m0 = mt * 16u, e0 = ns * 64u;
  const u32 g = lane >> 4, ln = lane & 15;

  const _Float16* Ab = Oc + m0 * (u32)Ee;
  const _Float16* Bb = WoH + e0 * (u32)Ee;

  v8f C[4] = {};
  for (u32 k0 = 0; k0 < (u32)Ee; k0 += 32) {
    v16h A = load_a_tile(Ab + k0, Ee, lane);
#pragma unroll
    for (int c = 0; c < 4; ++c) {
      v16h B = load_b_tile(Bb + (16u * c) * (u32)Ee + k0, Ee, lane);
      C[c] = WMMA16(A, B, C[c]);
    }
  }
  float* outb = out + (m0 + 8u * g) * (u32)Ee + e0 + ln;
#pragma unroll
  for (int c = 0; c < 4; ++c) {
    const float b = bo[e0 + 16u * c + ln];
#pragma unroll
    for (int r = 0; r < 8; ++r) {
      outb[(u32)r * (u32)Ee + (u32)(16 * c)] = C[c][r] + b;
    }
  }
}

// ---- host launcher ----------------------------------------------------------
extern "C" void kernel_launch(void* const* d_in, const int* in_sizes, int n_in,
                              void* d_out, int out_size, void* d_ws, size_t ws_size,
                              hipStream_t stream) {
  const float* values = (const float*)d_in[0];
  const float* keys   = (const float*)d_in[1];
  const float* query  = (const float*)d_in[2];
  const int*   mask   = (const int*)d_in[3];
  const float* Wv     = (const float*)d_in[4];
  const float* Wk     = (const float*)d_in[5];
  const float* Wq     = (const float*)d_in[6];
  const float* Wo     = (const float*)d_in[7];
  const float* bo     = (const float*)d_in[8];
  float* out = (float*)d_out;

  char* ws = (char*)d_ws;
  const size_t szProj = (size_t)Nn * Hh * Ll * Dd * sizeof(_Float16);  // 8 MB
  _Float16* Qh  = (_Float16*)ws;                 ws += szProj;
  _Float16* Kh  = (_Float16*)ws;                 ws += szProj;
  _Float16* Vt  = (_Float16*)ws;                 ws += szProj;
  _Float16* Ocw = (_Float16*)ws;                 ws += (size_t)Nn * Ll * Ee * sizeof(_Float16);
  _Float16* WoH = (_Float16*)ws;                 ws += (size_t)Ee * Ee * sizeof(_Float16);

  cvt_f16_kernel<<<(Ee * Ee + 255) / 256, 256, 0, stream>>>(Wo, WoH, Ee * Ee);
  proj_kernel<0><<<128, 128, 0, stream>>>(query, Wq, Qh);
  proj_kernel<1><<<128, 128, 0, stream>>>(keys, Wk, Kh);
  proj_kernel<2><<<128, 128, 0, stream>>>(values, Wv, Vt);
  attn_kernel<<<1024, 128, 0, stream>>>(Qh, Kh, Vt, mask, Ocw);
  outproj_kernel<<<1024, 128, 0, stream>>>(Ocw, WoH, bo, out);
}
